// Attention_28570122453652
// MI455X (gfx1250) — matrix operand order repaired and checked by reading, not compiled, not run on hardware
//
#include <hip/hip_runtime.h>

typedef _Float16 h16;
typedef _Float16 v16h __attribute__((ext_vector_type(16)));
typedef _Float16 v8h  __attribute__((ext_vector_type(8)));
typedef float    v8f  __attribute__((ext_vector_type(8)));

static constexpr int B_ = 4, S_ = 2048, D_ = 1024, H_ = 16, Dh_ = 64;

#if __has_builtin(__builtin_amdgcn_global_load_async_to_lds_b128)
#define HAVE_ASYNC_LDS 1
// builtin expects: (v4i __device__*, v4i __shared__*, int offset, int cpol)
// where v4i is a gcc-style vector_size(16) int.
typedef int v4i __attribute__((vector_size(16)));
typedef __attribute__((address_space(1))) v4i as1_v4i;
typedef __attribute__((address_space(3))) v4i as3_v4i;
#define STAGE_COPY(src_, dst_)                                                \
  __builtin_amdgcn_global_load_async_to_lds_b128((as1_v4i*)(src_),            \
                                                 (as3_v4i*)(dst_), 0, 0)
#else
#define HAVE_ASYNC_LDS 0
#define STAGE_COPY(src_, dst_) (*(v8h*)(dst_) = *(const v8h*)(src_))
#endif

__device__ __forceinline__ void wait_async0() {
#if HAVE_ASYNC_LDS
#if __has_builtin(__builtin_amdgcn_s_wait_asynccnt)
  __builtin_amdgcn_s_wait_asynccnt(0);
#else
  asm volatile("s_wait_asynccnt 0x0" ::: "memory");
#endif
#endif
}

__device__ __forceinline__ v8f vzero8() {
  v8f z = {0.f, 0.f, 0.f, 0.f, 0.f, 0.f, 0.f, 0.f};
  return z;
}

__device__ __forceinline__ v8f wmma16(v16h a, v16h b, v8f c) {
  // D = A(16x32 f16) * B(32x16 f16) + C(16x16 f32)
  return __builtin_amdgcn_wmma_f32_16x16x32_f16(false, a, false, b, (short)0, c,
                                                false, false);
}

// 16 contiguous f16 via two 16B loads (works for LDS rows with 16B alignment)
__device__ __forceinline__ v16h load_frag_pair(const h16* p) {
  v8h lo = *(const v8h*)p;
  v8h hi = *(const v8h*)(p + 8);
  v16h r;
#pragma unroll
  for (int i = 0; i < 8; ++i) { r[i] = lo[i]; r[8 + i] = hi[i]; }
  return r;
}

// B fragment: lane -> column n = l16, K = hf*16 + 0..15 contiguous (32B aligned)
__device__ __forceinline__ v16h load_b_frag(const h16* p) {
  return *(const v16h*)p;
}

// A fragment: lane row = l16; elems 0..15 -> K=k0+hf*16+i contiguous
// (two 16B chunks)
__device__ __forceinline__ v16h load_a_frag_h(const h16* row, int k0, int hf) {
  v8h lo = *(const v8h*)(row + k0 + hf * 16);
  v8h hi = *(const v8h*)(row + k0 + hf * 16 + 8);
  v16h r;
#pragma unroll
  for (int i = 0; i < 8; ++i) { r[i] = lo[i]; r[8 + i] = hi[i]; }
  return r;
}

// xor-shuffle reductions stay within each 16-lane half (C-layout halves)
__device__ __forceinline__ float rmax16(float v) {
#pragma unroll
  for (int m = 1; m < 16; m <<= 1) v = fmaxf(v, __shfl_xor(v, m, 32));
  return v;
}
__device__ __forceinline__ float rsum16(float v) {
#pragma unroll
  for (int m = 1; m < 16; m <<= 1) v += __shfl_xor(v, m, 32);
  return v;
}

// ---------------- elementwise f32 -> f16 ----------------
__global__ void cvt_f16_kernel(const float* __restrict__ src,
                               h16* __restrict__ dst, int n) {
  int i = blockIdx.x * blockDim.x + threadIdx.x;
  if (i < n) dst[i] = (h16)src[i];
}

// ---------------- weight transpose + f32 -> f16 convert ----------------
// w: [rows, cols] row-major f32  ->  wT: [cols, rows] row-major f16
__global__ void cvt_transpose_kernel(const float* __restrict__ w,
                                     h16* __restrict__ wT, int rows, int cols) {
  int idx = blockIdx.x * blockDim.x + threadIdx.x;
  if (idx >= rows * cols) return;
  int k = idx / cols;
  int n = idx - k * cols;
  wT[(size_t)n * rows + k] = (h16)w[idx];
}

// LDS B-tile: 64 cols x 32 K, row stride padded to 40 halfs (80B) so b128
// fragment reads hit 16 distinct bank groups. Double buffered: 2 x 5 KB.
#define BSM_STRIDE 40

// cooperative stage of one 64x32 B tile: 256 chunks of 16B, 2 per thread
#define STAGE(buf, k0)                                                        \
  {                                                                           \
    _Pragma("unroll") for (int i_ = 0; i_ < 2; ++i_) {                        \
      const int c_ = tid + i_ * 128;                                          \
      const int row_ = c_ >> 2;                                               \
      const int off_ = (c_ & 3) * 8;                                          \
      const h16* src_ = wT + (size_t)(nbase + row_) * D_ + (k0) + off_;       \
      h16* dst_ = &bsm[buf][row_ * BSM_STRIDE + off_];                        \
      STAGE_COPY(src_, dst_);                                                 \
    }                                                                         \
  }

// ---------------- fused QKV projection GEMM ----------------
// M=8192, N=3072, K=1024. Block = 4 waves, tile 128(M) x 64(N); wave: 32x64.
__global__ __launch_bounds__(128) void qkv_gemm_kernel(
    const h16* __restrict__ xh, const h16* __restrict__ wT,
    const float* __restrict__ bias, h16* __restrict__ qb,
    h16* __restrict__ kb, h16* __restrict__ vbT) {
  const int tid = threadIdx.x;
  const int wave = tid >> 5, lane = tid & 31;
  const int hf = lane >> 4, l16 = lane & 15;
  const int mbase = blockIdx.x * 128 + wave * 32;
  const int nbase = blockIdx.y * 64;

  __shared__ __align__(16) h16 bsm[2][64 * BSM_STRIDE];

  const h16* arow0 = xh + (size_t)(mbase + l16) * D_;
  const h16* arow1 = xh + (size_t)(mbase + 16 + l16) * D_;

  v8f acc[8];
#pragma unroll
  for (int t = 0; t < 8; ++t) acc[t] = vzero8();

  STAGE(0, 0)
  for (int k0 = 0; k0 < D_; k0 += 32) {
    const int cur = (k0 >> 5) & 1;
    wait_async0();       // this wave's copies for `cur` are done
    __syncthreads();     // all waves' copies visible; all prior reads retired
    if (k0 + 32 < D_) STAGE(cur ^ 1, k0 + 32)  // overlaps with WMMAs below
    v16h a0 = load_a_frag_h(arow0, k0, hf);
    v16h a1 = load_a_frag_h(arow1, k0, hf);
#pragma unroll
    for (int t = 0; t < 4; ++t) {
      v16h bf = load_frag_pair(&bsm[cur][(t * 16 + l16) * BSM_STRIDE + hf * 16]);
      acc[t] = wmma16(a0, bf, acc[t]);
      acc[4 + t] = wmma16(a1, bf, acc[4 + t]);
    }
  }

  // C layout: lane holds rows m = mg + hf*8 + r, col n = nbase + t*16 + l16
#pragma unroll
  for (int t = 0; t < 4; ++t) {
    const int n = nbase + t * 16 + l16;
    const float bn = bias[n];
    const int which = n >> 10;  // 0=Q 1=K 2=V
    const int nn = n & (D_ - 1);
    const int h = nn >> 6;
    const int d = nn & (Dh_ - 1);
#pragma unroll
    for (int g = 0; g < 2; ++g) {
      const int m0 = mbase + g * 16 + hf * 8;
      const int bidx = m0 >> 11;
      const int s0 = m0 & (S_ - 1);
      const v8f a = acc[g * 4 + t];
      if (which == 0) {
        h16* q = qb + ((size_t)(bidx * H_ + h) * S_ + s0) * Dh_ + d;
#pragma unroll
        for (int r = 0; r < 8; ++r)
          q[(size_t)r * Dh_] = (h16)((a[r] + bn) * 0.125f);  // 1/sqrt(Dh)
      } else if (which == 1) {
        h16* k = kb + ((size_t)(bidx * H_ + h) * S_ + s0) * Dh_ + d;
#pragma unroll
        for (int r = 0; r < 8; ++r) k[(size_t)r * Dh_] = (h16)(a[r] + bn);
      } else {  // V transposed [Dh, S]: contiguous along s -> one 16B store
        v8h vv;
#pragma unroll
        for (int r = 0; r < 8; ++r) vv[r] = (h16)(a[r] + bn);
        *(v8h*)(vbT + ((size_t)(bidx * H_ + h) * Dh_ + d) * S_ + s0) = vv;
      }
    }
  }
}

// ---------------- causal flash attention, one wave per (b,h,q-tile) ----------------
__global__ __launch_bounds__(32) void attn_kernel(const h16* __restrict__ qb,
                                                  const h16* __restrict__ kb,
                                                  const h16* __restrict__ vbT,
                                                  h16* __restrict__ ab) {
  const int lane = threadIdx.x & 31;
  const int hf = lane >> 4;
  const int l16 = lane & 15;
  const int q0 = blockIdx.x * 16;
  const int h = blockIdx.y;
  const int b = blockIdx.z;

  const h16* qh = qb + (size_t)(b * H_ + h) * S_ * Dh_;
  const h16* kh = kb + (size_t)(b * H_ + h) * S_ * Dh_;
  const h16* vh = vbT + (size_t)(b * H_ + h) * Dh_ * S_;

  // Q fragments resident: Dh=64 -> two K=32 fragments
  v16h qf0 = load_a_frag_h(qh + (size_t)(q0 + l16) * Dh_, 0, hf);
  v16h qf1 = load_a_frag_h(qh + (size_t)(q0 + l16) * Dh_, 32, hf);

  float m_run[8], l_run[8];
  v8f outacc[4];
#pragma unroll
  for (int r = 0; r < 8; ++r) { m_run[r] = -1e30f; l_run[r] = 0.0f; }
#pragma unroll
  for (int t = 0; t < 4; ++t) outacc[t] = vzero8();

  __shared__ __align__(32) h16 plds[16 * 32];  // probs D-layout -> A-layout

  for (int key0 = 0; key0 < q0 + 16; key0 += 32) {
    v8f s0 = vzero8(), s1 = vzero8();
    s0 = wmma16(qf0, load_b_frag(kh + (size_t)(key0 + l16) * Dh_ + hf * 16), s0);
    s0 = wmma16(qf1, load_b_frag(kh + (size_t)(key0 + l16) * Dh_ + 32 + hf * 16), s0);
    s1 = wmma16(qf0, load_b_frag(kh + (size_t)(key0 + 16 + l16) * Dh_ + hf * 16), s1);
    s1 = wmma16(qf1, load_b_frag(kh + (size_t)(key0 + 16 + l16) * Dh_ + 32 + hf * 16), s1);

#pragma unroll
    for (int r = 0; r < 8; ++r) {
      const int qrow = q0 + hf * 8 + r;
      float va = (key0 + l16 <= qrow) ? s0[r] : -10000.0f;  // causal mask
      float vb = (key0 + 16 + l16 <= qrow) ? s1[r] : -10000.0f;
      float mn = fmaxf(m_run[r], rmax16(fmaxf(va, vb)));
      float corr = __expf(m_run[r] - mn);
      m_run[r] = mn;
      float pa = __expf(va - mn);
      float pb = __expf(vb - mn);
      l_run[r] = l_run[r] * corr + rsum16(pa + pb);
#pragma unroll
      for (int t = 0; t < 4; ++t) outacc[t][r] *= corr;
      plds[(hf * 8 + r) * 32 + l16] = (h16)pa;
      plds[(hf * 8 + r) * 32 + 16 + l16] = (h16)pb;
    }
    __syncthreads();  // single-wave block: orders LDS
    v16h pf = load_a_frag_h(plds + (size_t)l16 * 32, 0, hf);
    __syncthreads();

    // PV: B from V^T [Dh,S] -> contiguous key run per lane
#pragma unroll
    for (int t = 0; t < 4; ++t)
      outacc[t] = wmma16(
          pf, load_b_frag(vh + (size_t)(t * 16 + l16) * S_ + key0 + hf * 16),
          outacc[t]);
  }

#pragma unroll
  for (int t = 0; t < 4; ++t) {
#pragma unroll
    for (int r = 0; r < 8; ++r) {
      const int qrow = q0 + hf * 8 + r;
      ab[(size_t)(b * S_ + qrow) * D_ + h * Dh_ + t * 16 + l16] =
          (h16)(outacc[t][r] / l_run[r]);
    }
  }
}

// ---------------- output projection GEMM (f32 out + bias) ----------------
// M=8192, N=1024, K=1024. Same tiling/pipeline as QKV GEMM.
__global__ __launch_bounds__(128) void proj_gemm_kernel(
    const h16* __restrict__ ab, const h16* __restrict__ wT,
    const float* __restrict__ bias, float* __restrict__ out) {
  const int tid = threadIdx.x;
  const int wave = tid >> 5, lane = tid & 31;
  const int hf = lane >> 4, l16 = lane & 15;
  const int mbase = blockIdx.x * 128 + wave * 32;
  const int nbase = blockIdx.y * 64;

  __shared__ __align__(16) h16 bsm[2][64 * BSM_STRIDE];

  const h16* arow0 = ab + (size_t)(mbase + l16) * D_;
  const h16* arow1 = ab + (size_t)(mbase + 16 + l16) * D_;

  v8f acc[8];
#pragma unroll
  for (int t = 0; t < 8; ++t) acc[t] = vzero8();

  STAGE(0, 0)
  for (int k0 = 0; k0 < D_; k0 += 32) {
    const int cur = (k0 >> 5) & 1;
    wait_async0();
    __syncthreads();
    if (k0 + 32 < D_) STAGE(cur ^ 1, k0 + 32)
    v16h a0 = load_a_frag_h(arow0, k0, hf);
    v16h a1 = load_a_frag_h(arow1, k0, hf);
#pragma unroll
    for (int t = 0; t < 4; ++t) {
      v16h bf = load_frag_pair(&bsm[cur][(t * 16 + l16) * BSM_STRIDE + hf * 16]);
      acc[t] = wmma16(a0, bf, acc[t]);
      acc[4 + t] = wmma16(a1, bf, acc[4 + t]);
    }
  }

#pragma unroll
  for (int t = 0; t < 4; ++t) {
    const int n = nbase + t * 16 + l16;
    const float bn = bias[n];
#pragma unroll
    for (int g = 0; g < 2; ++g) {
      const int m0 = mbase + g * 16 + hf * 8;
      const v8f a = acc[g * 4 + t];
#pragma unroll
      for (int r = 0; r < 8; ++r) out[(size_t)(m0 + r) * D_ + n] = a[r] + bn;
    }
  }
}

extern "C" void kernel_launch(void* const* d_in, const int* in_sizes, int n_in,
                              void* d_out, int out_size, void* d_ws, size_t ws_size,
                              hipStream_t stream) {
  const float* x        = (const float*)d_in[0];  // [4,2048,1024]
  const float* c_attn_w = (const float*)d_in[1];  // [1024,3072]
  const float* c_attn_b = (const float*)d_in[2];  // [3072]
  const float* c_proj_w = (const float*)d_in[3];  // [1024,1024]
  const float* c_proj_b = (const float*)d_in[4];  // [1024]
  float* out = (float*)d_out;

  char* ws = (char*)d_ws;
  h16* wqkvT  = (h16*)(ws);                  //  6 MB  [3072,1024] f16
  h16* wprojT = (h16*)(ws + 6291456);        //  2 MB  [1024,1024] f16
  h16* xh  = (h16*)(ws + 8388608);           // 16 MB  [B*S, D]    f16
  h16* qb  = (h16*)(ws + 25165824);          // 16 MB  [B,H,S,Dh]  (pre-scaled)
  h16* kb  = (h16*)(ws + 41943040);          // 16 MB  [B,H,S,Dh]
  h16* vbT = (h16*)(ws + 58720256);          // 16 MB  [B,H,Dh,S]
  h16* ab  = (h16*)(ws + 75497472);          // 16 MB  [B*S,D]

  cvt_f16_kernel<<<(8192 * 1024) / 256, 256, 0, stream>>>(x, xh, 8192 * 1024);
  cvt_transpose_kernel<<<(1024 * 3072) / 256, 256, 0, stream>>>(c_attn_w, wqkvT,
                                                                1024, 3072);
  cvt_transpose_kernel<<<(1024 * 1024) / 256, 256, 0, stream>>>(c_proj_w, wprojT,
                                                                1024, 1024);
  qkv_gemm_kernel<<<dim3(64, 48), 128, 0, stream>>>(xh, wqkvT, c_attn_b, qb, kb,
                                                    vbT);
  attn_kernel<<<dim3(128, 16, 4), 32, 0, stream>>>(qb, kb, vbT, ab);
  proj_gemm_kernel<<<dim3(64, 16), 128, 0, stream>>>(ab, wprojT, c_proj_b, out);
}
